// PsiCMPSCell_11072425689876
// MI455X (gfx1250) — matrix operand order repaired and checked by reading, not compiled, not run on hardware
//
#include <hip/hip_runtime.h>

// Problem constants (match reference)
#define T_LEN 4096
#define D_DIM 32
#define B_N   256
#define DT    0.001f

typedef __attribute__((ext_vector_type(16))) _Float16 v16h;
typedef __attribute__((ext_vector_type(8)))  float    v8f;

// ---------------------------------------------------------------------------
// Kernel 1: compute w[t] = A0 * (2/D) * dt * (c^T Rx c + s^T Rx s + c^T Ry s - s^T Ry c)
// (diagonal of R zeroed in the B fragments), one 16-t tile per wave via WMMA.
// ---------------------------------------------------------------------------
__global__ __launch_bounds__(256)
void psi_w_kernel(const float* __restrict__ Rx, const float* __restrict__ Ry,
                  const float* __restrict__ freqs, const float* __restrict__ A,
                  float* __restrict__ w)
{
    const int lane = threadIdx.x & 31;
    const int tile = blockIdx.x * (blockDim.x >> 5) + (threadIdx.x >> 5); // 0..255
    if (tile >= T_LEN / 16) return;   // uniform per wave

    const int half = lane >> 4;   // 0 or 1 (lane half)
    const int m    = lane & 15;   // row (A) / col (B,D) index

    // ---- B fragments: Rx^T and +/-Ry^T, 32x16 f16, two N-tiles ----
    // Layout (contiguous k, same order as the A fragments): lane-half selects
    // K group (0-15 / 16-31), element e -> K = 16*half + e, N = m (+16 for tile 1).
    v16h bx0, bx1, by0, by1, bny0, bny1;
#pragma unroll
    for (int e = 0; e < 16; ++e) {
        const int k  = 16 * half + e;
        const int n0 = m;
        const int n1 = m + 16;
        float rx0 = Rx[n0 * D_DIM + k];
        float rx1 = Rx[n1 * D_DIM + k];
        float ry0 = Ry[n0 * D_DIM + k];
        float ry1 = Ry[n1 * D_DIM + k];
        if (k == n0) { rx0 = 0.f; ry0 = 0.f; }   // remove R diagonal
        if (k == n1) { rx1 = 0.f; ry1 = 0.f; }
        bx0[e]  = (_Float16)rx0;   bx1[e]  = (_Float16)rx1;
        by0[e]  = (_Float16)ry0;   by1[e]  = (_Float16)ry1;
        bny0[e] = (_Float16)(-ry0); bny1[e] = (_Float16)(-ry1);
    }

    // ---- A fragments: Ct[t,k]=cos(t*f_k), St[t,k]=sin(t*f_k), 16x32 f16 ----
    // A layout (matches B): row M = lane%16; lanes 0-15 hold K=0..15,
    // lanes 16-31 hold K=16..31 (element e -> K = 16*half + e).
    const float tv_base = (float)(tile * 16);
    const float tval    = (tv_base + (float)m) * DT;
    v16h ac, as_;
#pragma unroll
    for (int e = 0; e < 16; ++e) {
        const int k = 16 * half + e;
        float s, c;
        __sincosf(tval * freqs[k], &s, &c);
        ac[e]  = (_Float16)c;
        as_[e] = (_Float16)s;
    }

    const v8f z = {0.f, 0.f, 0.f, 0.f, 0.f, 0.f, 0.f, 0.f};

    // acc1 = Ct @ Rx^T + St @ Ry^T   (chained through the accumulator)
    v8f acc1_0 = __builtin_amdgcn_wmma_f32_16x16x32_f16(false, ac,  false, bx0, (short)0, z,      false, false);
    acc1_0     = __builtin_amdgcn_wmma_f32_16x16x32_f16(false, as_, false, by0, (short)0, acc1_0, false, false);
    v8f acc1_1 = __builtin_amdgcn_wmma_f32_16x16x32_f16(false, ac,  false, bx1, (short)0, z,      false, false);
    acc1_1     = __builtin_amdgcn_wmma_f32_16x16x32_f16(false, as_, false, by1, (short)0, acc1_1, false, false);
    // acc2 = St @ Rx^T + Ct @ (-Ry)^T
    v8f acc2_0 = __builtin_amdgcn_wmma_f32_16x16x32_f16(false, as_, false, bx0,  (short)0, z,      false, false);
    acc2_0     = __builtin_amdgcn_wmma_f32_16x16x32_f16(false, ac,  false, bny0, (short)0, acc2_0, false, false);
    v8f acc2_1 = __builtin_amdgcn_wmma_f32_16x16x32_f16(false, as_, false, bx1,  (short)0, z,      false, false);
    acc2_1     = __builtin_amdgcn_wmma_f32_16x16x32_f16(false, ac,  false, bny1, (short)0, acc2_1, false, false);

    // ---- Combine: g_t = sum_a c_a*acc1[t,a] + s_a*acc2[t,a] ----
    // D layout: VGPR r -> row M = r (lanes 0-15) / r+8 (lanes 16-31), N = m.
    const float scale = A[0] * DT / 16.0f;   // A0 * 2 * dt / D
#pragma unroll
    for (int r = 0; r < 8; ++r) {
        const int   trow = r + 8 * half;
        const float tvr  = (tv_base + (float)trow) * DT;
        float s0, c0, s1, c1;
        __sincosf(tvr * freqs[m],      &s0, &c0);
        __sincosf(tvr * freqs[m + 16], &s1, &c1);
        float part = acc1_0[r] * c0 + acc2_0[r] * s0
                   + acc1_1[r] * c1 + acc2_1[r] * s1;
        // reduce across the 16 lanes of this half (xor masks stay in-half)
        part += __shfl_xor(part, 1, 32);
        part += __shfl_xor(part, 2, 32);
        part += __shfl_xor(part, 4, 32);
        part += __shfl_xor(part, 8, 32);
        if (m == 0)
            w[tile * 16 + trow] = scale * part;
    }
}

// ---------------------------------------------------------------------------
// Kernel 2: out[b,t] = x_seq[b,t,0] + w[t]  (pure HBM-bound stream)
// ---------------------------------------------------------------------------
__global__ __launch_bounds__(256)
void psi_apply_kernel(const float* __restrict__ x_seq, const float* __restrict__ w,
                      float* __restrict__ out)
{
    const int i4   = blockIdx.x * blockDim.x + threadIdx.x;
    const int base = i4 * 4;                       // 4 outputs per thread
    if (base >= B_N * T_LEN) return;
    const int t = base & (T_LEN - 1);              // T is power of two; t%4==0
    const float4 w4 = *reinterpret_cast<const float4*>(w + t);
    const float4 x0 = *reinterpret_cast<const float4*>(x_seq + 2 * base);
    const float4 x1 = *reinterpret_cast<const float4*>(x_seq + 2 * base + 4);
    float4 o;
    o.x = x0.x + w4.x;   // x_seq[...,0] is the even lane of each (val,time) pair
    o.y = x0.z + w4.y;
    o.z = x1.x + w4.z;
    o.w = x1.z + w4.w;
    *reinterpret_cast<float4*>(out + base) = o;
}

extern "C" void kernel_launch(void* const* d_in, const int* in_sizes, int n_in,
                              void* d_out, int out_size, void* d_ws, size_t ws_size,
                              hipStream_t stream) {
    (void)in_sizes; (void)n_in; (void)out_size; (void)ws_size;
    const float* x_seq = (const float*)d_in[0];   // (B, T, 2)
    const float* A     = (const float*)d_in[1];   // (1,)
    const float* Rx    = (const float*)d_in[2];   // (D, D)
    const float* Ry    = (const float*)d_in[3];   // (D, D)
    const float* freqs = (const float*)d_in[4];   // (D,)
    // d_in[5], d_in[6] (psi_x, psi_y): dead w.r.t. the reference output.

    float* w   = (float*)d_ws;                    // 4096 floats of scratch
    float* out = (float*)d_out;                   // (B, T)

    // 256 tiles of 16 t-values, one tile per wave: 32 blocks * 8 waves.
    psi_w_kernel<<<32, 256, 0, stream>>>(Rx, Ry, freqs, A, w);

    const int total4 = (B_N * T_LEN) / 4;         // 262144 threads
    psi_apply_kernel<<<total4 / 256, 256, 0, stream>>>(x_seq, w, out);
}
